// CnnWordSeg_28853590295118
// MI455X (gfx1250) — matrix operand order repaired and checked by reading, not compiled, not run on hardware
//
#include <hip/hip_runtime.h>

// ---------------------------------------------------------------------------
// CNN-CRF word segmenter for MI455X (gfx1250, wave32).
//   B=64, S=2048, V=32000, H=128, T=4
// Conv stack: v_wmma_f32_16x16x32_bf16 (bf16 mul, f32 accum), LDS-staged
// activations, register-resident weight fragments, batched A-fragment loads.
// ---------------------------------------------------------------------------

#define B_    64
#define S_    2048
#define V_    32000
#define H_    128
#define T_    4
#define TILE  128                 // sequence positions per workgroup
#define HALO  8                   // staged halo rows each side (>=3 needed)
#define RROWS (TILE + 2*HALO)     // 144 staged rows
#define PITCH 136                 // bf16 elems per LDS row (128 + 8 pad)

typedef __attribute__((ext_vector_type(16))) __bf16         v16bf;
typedef __attribute__((ext_vector_type(8)))  float          v8f;
typedef __attribute__((ext_vector_type(8)))  unsigned short ushort8;
typedef __attribute__((ext_vector_type(2)))  float          f32x2;

union FragA { ushort8 u[2]; v16bf v; };
union FragB { unsigned short s[16]; v16bf v; };

struct F3 { float x, y, z; };     // one input channel, all 3 conv taps

// Native converts (gfx1250 hardware bf16 cvt; RNE by default).
__device__ __forceinline__ unsigned short f2bf(float f) {
    __bf16 h = (__bf16)f;
    return __builtin_bit_cast(unsigned short, h);
}
__device__ __forceinline__ float bf2f(unsigned short h) {
    return (float)__builtin_bit_cast(__bf16, h);
}
__device__ __forceinline__ unsigned int f2bf_pk(float lo, float hi) {
    return (unsigned int)f2bf(lo) | ((unsigned int)f2bf(hi) << 16);
}

// One conv layer: out[r, o] = relu( sum_{tap,i} in[r+tap-1, i] * W[o,i,tap] + bias[o] )
// Each wave owns a 16-wide output-channel column; B fragments stay in VGPRs.
__device__ __forceinline__ void conv_layer_wmma(
    const unsigned short* __restrict__ in,
    unsigned short*       __restrict__ out,
    const float*          __restrict__ W,    // (H, H, 3) : W[o*H*3 + i*3 + tap]
    const float*          __restrict__ Bv,   // (H,)
    int tile_base, int lane, int wave)
{
    const int o0    = wave * 16;
    const int oc    = o0 + (lane & 15);       // this lane's output channel (N)
    const int khalf = lane >> 4;              // upper/lower half-wave K split
    const float bias = Bv[oc];

    // ---- 12 register-resident B fragments: [tap][chunk], K = chunk*32 .. +31
    // B layout used here (contiguous K, not the ISA table order; 16-bit, 32x16): lane = N column; lanes 0-15 hold K 0..15,
    // lanes 16-31 hold K 16..31. One float3 load per input channel yields all
    // three taps at once.
    FragB bfrag[12];
#pragma unroll
    for (int ch = 0; ch < 4; ++ch) {
        FragB f0, f1, f2;
        const int kstart = ch * 32 + khalf * 16;
        const F3* wp = (const F3*)(W + (size_t)oc * (H_ * 3) + kstart * 3);
#pragma unroll
        for (int j = 0; j < 16; ++j) {
            const F3 w = wp[j];
            f0.s[j] = f2bf(w.x);
            f1.s[j] = f2bf(w.y);
            f2.s[j] = f2bf(w.z);
        }
        bfrag[0 * 4 + ch] = f0;
        bfrag[1 * 4 + ch] = f1;
        bfrag[2 * 4 + ch] = f2;
    }

    // Sequence-edge clamping only matters for the first/last tile of the grid.
    const bool interior = (tile_base > 0) && (tile_base + TILE < S_);

#pragma unroll 1
    for (int mt = 0; mt < RROWS / 16; ++mt) {
        const int rbase = mt * 16 + (lane & 15);          // A-fragment row (M)

        // Row pointers for the three taps (cheap clamp on interior tiles).
        const unsigned short* rowp[3];
#pragma unroll
        for (int tap = 0; tap < 3; ++tap) {
            int rin;
            if (interior) {
                rin = min(max(rbase + (tap - 1), 0), RROWS - 1);
            } else {
                int sp = tile_base + rbase + (tap - 1) - HALO;
                sp  = min(max(sp, 0), S_ - 1);
                rin = min(max(sp - tile_base + HALO, 0), RROWS - 1);
            }
            rowp[tap] = in + rin * PITCH;
        }

        // Batch ALL 24 ds_load_b128 for this tile before any WMMA: one dscnt
        // wait amortized over 12 WMMAs instead of a full LDS round-trip each.
        FragA a[12];
#pragma unroll
        for (int tap = 0; tap < 3; ++tap) {
#pragma unroll
            for (int ch = 0; ch < 4; ++ch) {
                // A layout matched to B: lanes 0-15: K 0..15,
                // lanes 16-31: K 16..31 -> two 16B LDS loads.
                const int c0 = ch * 32 + khalf * 16;
                a[tap * 4 + ch].u[0] = *(const ushort8*)(rowp[tap] + c0);
                a[tap * 4 + ch].u[1] = *(const ushort8*)(rowp[tap] + c0 + 8);
            }
        }

        // Two independent accumulator chains to avoid tail-to-head WMMA deps.
        v8f acc0 = {};
        v8f acc1 = {};
#pragma unroll
        for (int f = 0; f < 12; f += 2) {
            acc0 = __builtin_amdgcn_wmma_f32_16x16x32_bf16(
                    false, a[f].v, false, bfrag[f].v, (short)0, acc0, false, false);
            acc1 = __builtin_amdgcn_wmma_f32_16x16x32_bf16(
                    false, a[f + 1].v, false, bfrag[f + 1].v, (short)0, acc1, false, false);
        }

        // C layout: lane N = oc; VGPR r -> M = r (+8 for upper half-wave)
        const int mrow0 = mt * 16 + khalf * 8;
#pragma unroll
        for (int r = 0; r < 8; ++r) {
            const float v = fmaxf(acc0[r] + acc1[r] + bias, 0.0f);
            out[(mrow0 + r) * PITCH + oc] = f2bf(v);
        }
    }
}

__global__ __launch_bounds__(256) void fused_conv_emit_kernel(
    const int*   __restrict__ x,     // (B, S)
    const float* __restrict__ emb,   // (V, H)
    const float* __restrict__ w1, const float* __restrict__ b1,
    const float* __restrict__ w2, const float* __restrict__ b2,
    const float* __restrict__ w3, const float* __restrict__ b3,
    const float* __restrict__ dw,    // (H, T)
    const float* __restrict__ db,    // (T,)
    float*       __restrict__ E)     // (B, S, T) emissions out
{
    __shared__ __align__(16) unsigned short buf[2][RROWS * PITCH];
    __shared__ float dwsh[H_ * T_];

    const int tid  = threadIdx.x;
    const int lane = tid & 31;
    const int wave = tid >> 5;                 // 8 waves -> 8 N-columns
    const int b    = blockIdx.y;
    const int tile_base = blockIdx.x * TILE;

    // ---- stage dw (2 KB) for the emission pass
    for (int i = tid; i < H_ * T_; i += 256) dwsh[i] = dw[i];

    // ---- stage embeddings (gather, fp32 -> packed bf16) with edge clamping
    for (int idx = tid; idx < RROWS * (H_ / 2); idx += 256) {
        const int r  = idx >> 6;               // / (H_/2)
        const int c2 = (idx & 63) * 2;         // even channel
        int sp = tile_base + r - HALO;
        sp = min(max(sp, 0), S_ - 1);
        const int tok = x[b * S_ + sp];
        const f32x2 e2 = *(const f32x2*)(emb + (size_t)tok * H_ + c2);
        ((unsigned int*)buf[0])[(r * PITCH + c2) >> 1] = f2bf_pk(e2.x, e2.y);
    }
    __syncthreads();

    conv_layer_wmma(buf[0], buf[1], w1, b1, tile_base, lane, wave);
    __syncthreads();
    conv_layer_wmma(buf[1], buf[0], w2, b2, tile_base, lane, wave);
    __syncthreads();
    conv_layer_wmma(buf[0], buf[1], w3, b3, tile_base, lane, wave);
    __syncthreads();

    // ---- emissions: e[s,t] = h[s,:] . dw[:,t] + db[t]   (T=4, VALU)
    {
        const unsigned short* hbuf = buf[1];
        const int row = tid >> 1;                       // 0..127
        const int t0  = (tid & 1) * 2;                  // {0,2}
        const unsigned short* rowp = hbuf + (HALO + row) * PITCH;
        float a0 = db[t0], a1 = db[t0 + 1];
        for (int c8 = 0; c8 < H_; c8 += 8) {
            const ushort8 h8 = *(const ushort8*)(rowp + c8);
#pragma unroll
            for (int j = 0; j < 8; ++j) {
                const float hv = bf2f(h8[j]);
                a0 += hv * dwsh[(c8 + j) * T_ + t0];
                a1 += hv * dwsh[(c8 + j) * T_ + t0 + 1];
            }
        }
        const int s = tile_base + row;
        float* ep = E + ((size_t)b * S_ + s) * T_;
        ep[t0]     = a0;
        ep[t0 + 1] = a1;
    }
}

// ---------------------------------------------------------------------------
// CRF log-likelihood per batch: lanes 0..3 carry alpha[t]; logsumexp over the
// 4x4 transition via shuffles. One wave32 per batch, sequential over S.
// ---------------------------------------------------------------------------
__global__ __launch_bounds__(32) void crf_kernel(
    const float*         __restrict__ E,      // (B, S, T)
    const int*           __restrict__ y,      // (B, S)
    const unsigned char* __restrict__ mask,   // (B, S) bool
    const float*         __restrict__ start_t,
    const float*         __restrict__ end_t,
    const float*         __restrict__ trans,  // (T, T)
    float*               __restrict__ part)   // (B,) per-batch llh
{
    const int b    = blockIdx.x;
    const int lane = threadIdx.x;

    float tr0 = 0.f, tr1 = 0.f, tr2 = 0.f, tr3 = 0.f;   // column `lane` of trans
    if (lane < T_) {
        tr0 = trans[0 * T_ + lane]; tr1 = trans[1 * T_ + lane];
        tr2 = trans[2 * T_ + lane]; tr3 = trans[3 * T_ + lane];
    }

    const float*         Eb = E + (size_t)b * S_ * T_;
    const int*           yb = y + b * S_;
    const unsigned char* mb = mask + b * S_;

    float alpha = -1e30f;
    if (lane < T_) alpha = start_t[lane] + Eb[lane];

    int   yprev = yb[0];
    float num   = start_t[yprev] + Eb[yprev];

    for (int s = 1; s < S_; ++s) {
        const int m = mb[s];
        const float a0 = __shfl(alpha, 0, 32);
        const float a1 = __shfl(alpha, 1, 32);
        const float a2 = __shfl(alpha, 2, 32);
        const float a3 = __shfl(alpha, 3, 32);
        if (lane < T_) {
            const float t0 = a0 + tr0, t1 = a1 + tr1;
            const float t2 = a2 + tr2, t3 = a3 + tr3;
            const float mx = fmaxf(fmaxf(t0, t1), fmaxf(t2, t3));
            const float lse = mx + __logf(__expf(t0 - mx) + __expf(t1 - mx) +
                                          __expf(t2 - mx) + __expf(t3 - mx));
            const float nxt = lse + Eb[s * T_ + lane];
            alpha = m ? nxt : alpha;
        }
        if (m) {
            const int yc = yb[s];
            num += trans[yprev * T_ + yc] + Eb[s * T_ + yc];
            yprev = yc;
        }
    }
    num += end_t[yprev];

    const float v  = (lane < T_) ? (alpha + end_t[lane]) : -1e30f;
    const float v0 = __shfl(v, 0, 32), v1 = __shfl(v, 1, 32);
    const float v2 = __shfl(v, 2, 32), v3 = __shfl(v, 3, 32);
    const float mx = fmaxf(fmaxf(v0, v1), fmaxf(v2, v3));
    const float den = mx + __logf(__expf(v0 - mx) + __expf(v1 - mx) +
                                  __expf(v2 - mx) + __expf(v3 - mx));
    if (lane == 0) part[b] = num - den;
}

// Deterministic in-order final reduction (no float atomics -> replay-stable).
__global__ void reduce_llh_kernel(const float* __restrict__ part,
                                  float* __restrict__ out)
{
    if (threadIdx.x == 0 && blockIdx.x == 0) {
        float s = 0.0f;
        for (int i = 0; i < B_; ++i) s += part[i];
        out[0] = s;
    }
}

extern "C" void kernel_launch(void* const* d_in, const int* in_sizes, int n_in,
                              void* d_out, int out_size, void* d_ws, size_t ws_size,
                              hipStream_t stream)
{
    (void)in_sizes; (void)n_in; (void)out_size; (void)ws_size;
    const int*           x       = (const int*)          d_in[0];
    const int*           y       = (const int*)          d_in[1];
    const unsigned char* mask    = (const unsigned char*)d_in[2];
    const float*         emb     = (const float*)        d_in[3];
    const float*         w1      = (const float*)        d_in[4];
    const float*         b1      = (const float*)        d_in[5];
    const float*         w2      = (const float*)        d_in[6];
    const float*         b2      = (const float*)        d_in[7];
    const float*         w3      = (const float*)        d_in[8];
    const float*         b3      = (const float*)        d_in[9];
    const float*         dw      = (const float*)        d_in[10];
    const float*         db      = (const float*)        d_in[11];
    const float*         start_t = (const float*)        d_in[12];
    const float*         end_t   = (const float*)        d_in[13];
    const float*         trans   = (const float*)        d_in[14];

    float* E    = (float*)d_ws;                       // B*S*T emissions (2 MB)
    float* part = E + (size_t)B_ * S_ * T_;           // B partial llh

    dim3 grid1(S_ / TILE, B_);
    fused_conv_emit_kernel<<<grid1, 256, 0, stream>>>(
        x, emb, w1, b1, w2, b2, w3, b3, dw, db, E);

    crf_kernel<<<dim3(B_), 32, 0, stream>>>(E, y, mask, start_t, end_t, trans, part);

    reduce_llh_kernel<<<1, 32, 0, stream>>>(part, (float*)d_out);
}
